// AR_62972810494409
// MI455X (gfx1250) — matrix operand order repaired and checked by reading, not compiled, not run on hardware
//
#include <hip/hip_runtime.h>
#include <hip/hip_bf16.h>

// ---------------------------------------------------------------------------
// Problem constants
// ---------------------------------------------------------------------------
#define DD      1024
#define HH      16
#define DH      64
#define NLAYERS 12
#define VV      1024
#define BB      2
#define TT      896           // 128 + 256 + 512
#define BT      (BB * TT)     // 1792
#define TCODE   512

typedef __attribute__((ext_vector_type(16))) __bf16 bf16x16;
typedef __attribute__((ext_vector_type(8)))  float  floatx8;

__device__ __forceinline__ __bf16 f2bf(float f) {
  unsigned u = __builtin_bit_cast(unsigned, f);
  unsigned r = u + 0x7FFFu + ((u >> 16) & 1u);   // round-to-nearest-even
  unsigned short s = (unsigned short)(r >> 16);
  return __builtin_bit_cast(__bf16, s);
}

// gfx1250 async memory->LDS copy (16B per lane), ASYNCcnt-tracked.
// VDST = per-lane LDS byte address, VADDR = per-lane 64-bit global address.
__device__ __forceinline__ void async_g2l_b128(const void* gaddr, void* lds) {
  unsigned laddr = (unsigned)(unsigned long long)lds;       // LDS offset bits
  unsigned long long ga = (unsigned long long)gaddr;
  asm volatile("global_load_async_to_lds_b128 %0, %1, off"
               :: "v"(laddr), "v"(ga) : "memory");
}
__device__ __forceinline__ void wait_async(int n) {
  if (n == 0) asm volatile("s_wait_asynccnt 0x0" ::: "memory");
  else        asm volatile("s_wait_asynccnt 0x4" ::: "memory");
}

// ---------------------------------------------------------------------------
// f32 -> bf16 conversion (weights)
// ---------------------------------------------------------------------------
__global__ void cvt_bf16_kernel(const float* __restrict__ in,
                                __bf16* __restrict__ out, long n) {
  long i = (long)blockIdx.x * blockDim.x + threadIdx.x;
  if (i < n) out[i] = f2bf(in[i]);
}

// ---------------------------------------------------------------------------
// Concatenate [text|prom|code] -> x (f32) and xb (bf16)
// ---------------------------------------------------------------------------
__global__ void concat_kernel(const float* __restrict__ text,
                              const float* __restrict__ prom,
                              const float* __restrict__ code,
                              float* __restrict__ x, __bf16* __restrict__ xb) {
  long idx = (long)blockIdx.x * blockDim.x + threadIdx.x;
  if (idx >= (long)BT * DD) return;
  int  c  = idx & (DD - 1);
  long rt = idx >> 10;
  int  t  = (int)(rt % TT);
  int  b  = (int)(rt / TT);
  float v;
  if (t < 128)      v = text[((long)b * 128 + t) * DD + c];
  else if (t < 384) v = prom[((long)b * 256 + (t - 128)) * DD + c];
  else              v = code[((long)b * 512 + (t - 384)) * DD + c];
  x[idx]  = v;
  xb[idx] = f2bf(v);
}

// ---------------------------------------------------------------------------
// Precompute the (layer-invariant) edited additive attention mask once,
// so the flash kernel's inner loop is pure load+fma instead of cmp/cndmask.
// ---------------------------------------------------------------------------
__global__ void build_amask_kernel(const float* __restrict__ mask,
                                   float* __restrict__ am) {
  long idx = (long)blockIdx.x * blockDim.x + threadIdx.x;
  if (idx >= (long)BB * TT * TT) return;
  int  kg = (int)(idx % TT);
  long r  = idx / TT;
  int  qg = (int)(r % TT);
  float m = mask[idx];
  if (qg < (TT - TCODE + 128) && kg >= (TT - TCODE)) m = 0.0f;          // q<512, k>=384
  if (qg >= (TT - TCODE) && kg >= (TT - TCODE) && kg > qg) m = 0.0f;    // tril
  am[idx] = m;
}

// ---------------------------------------------------------------------------
// WMMA bf16 GEMM:  C[M,N] = A[M,K] @ W[N,K]^T + bias
// Block tile 128x128, 8 waves (wave32), each wave 32x64 = 2x4 WMMA tiles.
// Double-buffered LDS staging via global_load_async_to_lds_b128:
// DMA of tile i+1 overlaps WMMA compute of tile i (ASYNCcnt pipeline).
// EPI: 0 = f32 out, 1 = bf16 out, 2 = exact-GELU then bf16 out
// ---------------------------------------------------------------------------
template <int EPI>
__global__ __launch_bounds__(256) void gemm_bf16_kernel(
    const __bf16* __restrict__ A, const __bf16* __restrict__ W,
    const float* __restrict__ bias,
    float* __restrict__ Cf, __bf16* __restrict__ Cb,
    int M, int N, int K) {
  __shared__ __align__(16) __bf16 As[2][128][32];
  __shared__ __align__(16) __bf16 Bs[2][128][32];

  const int tid  = threadIdx.x;
  const int lane = tid & 31;
  const int wave = tid >> 5;
  const int wm   = wave >> 1;      // 0..3 (M dir, 32 rows each)
  const int wn   = wave & 1;       // 0..1 (N dir, 64 cols each)
  const int lr   = lane & 15;
  const int hl   = lane >> 4;
  const long m0  = (long)blockIdx.y * 128;
  const long n0  = (long)blockIdx.x * 128;

  floatx8 acc[2][4] = {};

  // Staging: 128 rows x 32 cols bf16 per matrix; 256 threads x 2 x 16B each.
  const int rowL = tid >> 1;            // 0..127
  const int colL = (tid & 1) << 4;      // 0 or 16

  auto stage = [&](int buf, int k0) {
    const __bf16* ga0 = &A[(m0 + rowL) * (long)K + k0 + colL];
    const __bf16* gb0 = &W[(n0 + rowL) * (long)K + k0 + colL];
    async_g2l_b128(ga0,     &As[buf][rowL][colL]);
    async_g2l_b128(ga0 + 8, &As[buf][rowL][colL + 8]);
    async_g2l_b128(gb0,     &Bs[buf][rowL][colL]);
    async_g2l_b128(gb0 + 8, &Bs[buf][rowL][colL + 8]);
  };

  const int nk = K >> 5;
  stage(0, 0);
  for (int it = 0; it < nk; ++it) {
    const int buf = it & 1;
    if (it + 1 < nk) { stage(buf ^ 1, (it + 1) << 5); wait_async(4); }
    else             { wait_async(0); }
    __syncthreads();

    // A fragment: 16x32, half-wave K-split per ISA 7.12.2
    bf16x16 af[2], bw[4];
#pragma unroll
    for (int i = 0; i < 2; ++i) {
      const __bf16* ap = &As[buf][wm * 32 + i * 16 + lr][hl * 8];
      ((uint4*)&af[i])[0] = *(const uint4*)ap;          // K = hl*8 .. +7
      ((uint4*)&af[i])[1] = *(const uint4*)(ap + 16);   // K = 16+hl*8 .. +7
    }
    // B fragment: 32x16, lane = N column, K = hl*8..+7 and 16+hl*8..+7
#pragma unroll
    for (int j = 0; j < 4; ++j) {
      const __bf16* bp = &Bs[buf][wn * 64 + j * 16 + lr][hl * 8];
      ((uint4*)&bw[j])[0] = *(const uint4*)bp;
      ((uint4*)&bw[j])[1] = *(const uint4*)(bp + 16);
    }
#pragma unroll
    for (int i = 0; i < 2; ++i)
#pragma unroll
      for (int j = 0; j < 4; ++j)
        acc[i][j] = __builtin_amdgcn_wmma_f32_16x16x32_bf16(
            false, af[i], false, bw[j], (short)0, acc[i][j], false, false);
    __syncthreads();
  }

  // Epilogue: C layout — lane<16: (M=e, N=lane); lane>=16: (M=e+8, N=lane-16)
#pragma unroll
  for (int i = 0; i < 2; ++i) {
#pragma unroll
    for (int j = 0; j < 4; ++j) {
      const long col = n0 + wn * 64 + j * 16 + lr;
      const float bv = bias ? bias[col] : 0.0f;
#pragma unroll
      for (int e = 0; e < 8; ++e) {
        const long row = m0 + wm * 32 + i * 16 + hl * 8 + e;
        float v = acc[i][j][e] + bv;
        if (EPI == 2) v = 0.5f * v * (1.0f + erff(v * 0.70710678118654752440f));
        if (EPI == 0) Cf[row * (long)N + col] = v;
        else          Cb[row * (long)N + col] = f2bf(v);
      }
    }
  }
}

// ---------------------------------------------------------------------------
// Flash attention: one wave per (b, h, 16-query tile). K/V staged in LDS via
// async-to-LDS, scores via WMMA (q @ k^T), online softmax, P@V via WMMA.
// Mask is an ADDITIVE bias (precomputed, not -inf), so all k-tiles run.
// ---------------------------------------------------------------------------
__global__ __launch_bounds__(32) void attn_kernel(
    const __bf16* __restrict__ qkv,   // [BT, 3*DD] bf16 (q|k|v)
    const float* __restrict__ amask,  // [B, T, T] f32, pre-edited
    __bf16* __restrict__ o) {         // [BT, DD] bf16
  __shared__ __align__(16) __bf16 Ks[32][64];
  __shared__ __align__(16) __bf16 Vs[32][64];
  __shared__ __align__(16) __bf16 Ps[16][32];

  const int lane = threadIdx.x;
  const int lr = lane & 15, hl = lane >> 4;
  const int qt = blockIdx.x, h = blockIdx.y, b = blockIdx.z;
  const int q0 = qt * 16;

  // Q A-fragments for dh chunks [0..31], [32..63]
  bf16x16 aq[2];
#pragma unroll
  for (int c = 0; c < 2; ++c) {
    const __bf16* qp = qkv + (long)(b * TT + q0 + lr) * (3 * DD) + h * DH + c * 32;
    ((uint4*)&aq[c])[0] = *(const uint4*)(qp + hl * 8);
    ((uint4*)&aq[c])[1] = *(const uint4*)(qp + 16 + hl * 8);
  }

  floatx8 oacc[4] = {};
  float m_i[8], l_i[8];
#pragma unroll
  for (int e = 0; e < 8; ++e) { m_i[e] = -1e30f; l_i[e] = 0.0f; }
  const float scale = 0.125f;   // 1/sqrt(64)

  for (int kt = 0; kt < TT; kt += 32) {
    // Async-stage K and V rows [kt, kt+32) for this head into LDS
    {
      const __bf16* kp = qkv + (long)(b * TT + kt + lane) * (3 * DD) + DD + h * DH;
      const __bf16* vp = kp + DD;
#pragma unroll
      for (int c = 0; c < 8; ++c) {
        async_g2l_b128(kp + c * 8, &Ks[lane][c * 8]);
        async_g2l_b128(vp + c * 8, &Vs[lane][c * 8]);
      }
      wait_async(0);
    }
    __syncthreads();

    // scores tile 16x32: two 16x16 accums, K-dim = dh in two 32-chunks
    floatx8 sc[2] = {};
#pragma unroll
    for (int nt = 0; nt < 2; ++nt) {
#pragma unroll
      for (int c = 0; c < 2; ++c) {
        bf16x16 bk;
        const __bf16* bp = &Ks[nt * 16 + lr][c * 32 + hl * 8];
        ((uint4*)&bk)[0] = *(const uint4*)bp;
        ((uint4*)&bk)[1] = *(const uint4*)(bp + 16);
        sc[nt] = __builtin_amdgcn_wmma_f32_16x16x32_bf16(
            false, aq[c], false, bk, (short)0, sc[nt], false, false);
      }
    }

    // scale + additive mask bias (pure load + fma now)
    float sv[2][8], mx[8];
#pragma unroll
    for (int e = 0; e < 8; ++e) mx[e] = -1e30f;
#pragma unroll
    for (int nt = 0; nt < 2; ++nt) {
#pragma unroll
      for (int e = 0; e < 8; ++e) {
        int qg = q0 + hl * 8 + e;
        int kg = kt + nt * 16 + lr;
        float s = fmaf(sc[nt][e], scale, amask[((long)b * TT + qg) * TT + kg]);
        sv[nt][e] = s;
        mx[e] = fmaxf(mx[e], s);
      }
    }
    // row reductions across the 16-lane N groups (wave32 halves)
#pragma unroll
    for (int e = 0; e < 8; ++e) {
#pragma unroll
      for (int off = 8; off >= 1; off >>= 1)
        mx[e] = fmaxf(mx[e], __shfl_xor(mx[e], off, 32));
    }
    float corr[8], rs[8];
#pragma unroll
    for (int e = 0; e < 8; ++e) {
      float mn = fmaxf(m_i[e], mx[e]);
      corr[e] = __expf(m_i[e] - mn);
      m_i[e] = mn;
      rs[e] = 0.0f;
    }
#pragma unroll
    for (int nt = 0; nt < 2; ++nt) {
#pragma unroll
      for (int e = 0; e < 8; ++e) {
        float p = __expf(sv[nt][e] - m_i[e]);
        rs[e] += p;
        Ps[hl * 8 + e][nt * 16 + lr] = f2bf(p);   // LDS roundtrip -> A layout
      }
    }
#pragma unroll
    for (int e = 0; e < 8; ++e) {
#pragma unroll
      for (int off = 8; off >= 1; off >>= 1)
        rs[e] += __shfl_xor(rs[e], off, 32);
      l_i[e] = l_i[e] * corr[e] + rs[e];
#pragma unroll
      for (int n = 0; n < 4; ++n) oacc[n][e] *= corr[e];
    }
    __syncthreads();

    // P fragment (16x32 A layout)
    bf16x16 ap;
    {
      const __bf16* pp = &Ps[lr][hl * 16];
      ((uint4*)&ap)[0] = *(const uint4*)pp;
      ((uint4*)&ap)[1] = *(const uint4*)(pp + 8);
    }
    // O += P @ V : 4 WMMA tiles along dh
#pragma unroll
    for (int n = 0; n < 4; ++n) {
      bf16x16 bv;
#pragma unroll
      for (int j = 0; j < 16; ++j)
        bv[j] = Vs[hl * 16 + j][n * 16 + lr];
      oacc[n] = __builtin_amdgcn_wmma_f32_16x16x32_bf16(
          false, ap, false, bv, (short)0, oacc[n], false, false);
    }
    __syncthreads();
  }

  // normalize + store bf16
#pragma unroll
  for (int n = 0; n < 4; ++n) {
#pragma unroll
    for (int e = 0; e < 8; ++e) {
      int qg = q0 + hl * 8 + e;
      float v = oacc[n][e] / l_i[e];
      o[(long)(b * TT + qg) * DD + h * DH + n * 16 + lr] = f2bf(v);
    }
  }
}

// ---------------------------------------------------------------------------
// x = LN(x + delta) * g + beta ; writes f32 x and bf16 xb
// ---------------------------------------------------------------------------
__global__ __launch_bounds__(256) void add_ln_kernel(
    float* __restrict__ x, const float* __restrict__ delta,
    const float* __restrict__ g, const float* __restrict__ bta,
    __bf16* __restrict__ xb) {
  __shared__ float red[256];
  const int row = blockIdx.x, tid = threadIdx.x;
  float v[4], s = 0.0f;
#pragma unroll
  for (int i = 0; i < 4; ++i) {
    int c = i * 256 + tid;
    v[i] = x[(long)row * DD + c] + delta[(long)row * DD + c];
    s += v[i];
  }
  red[tid] = s; __syncthreads();
  for (int o = 128; o > 0; o >>= 1) { if (tid < o) red[tid] += red[tid + o]; __syncthreads(); }
  float mu = red[0] * (1.0f / DD);
  __syncthreads();
  float s2 = 0.0f;
#pragma unroll
  for (int i = 0; i < 4; ++i) { float d = v[i] - mu; s2 += d * d; }
  red[tid] = s2; __syncthreads();
  for (int o = 128; o > 0; o >>= 1) { if (tid < o) red[tid] += red[tid + o]; __syncthreads(); }
  float rstd = rsqrtf(red[0] * (1.0f / DD) + 1e-5f);
#pragma unroll
  for (int i = 0; i < 4; ++i) {
    int c = i * 256 + tid;
    float y = (v[i] - mu) * rstd * g[c] + bta[c];
    x[(long)row * DD + c]  = y;
    xb[(long)row * DD + c] = f2bf(y);
  }
}

// ---------------------------------------------------------------------------
// Loss: per selected row log-softmax NLL, atomically accumulated.
// ---------------------------------------------------------------------------
__global__ void zero2_kernel(float* acc) { if (threadIdx.x < 2) acc[threadIdx.x] = 0.0f; }

__global__ __launch_bounds__(256) void loss_kernel(
    const float* __restrict__ logits,              // [B*512, V]
    const int* __restrict__ gt,                    // [B,512]
    const unsigned char* __restrict__ gt_mask,     // [B,512] bool bytes
    float* __restrict__ acc) {
  __shared__ float red[256];
  const int row = blockIdx.x, tid = threadIdx.x;
  const float* lp = logits + (long)row * VV;
  float mx = -1e30f;
#pragma unroll
  for (int i = 0; i < 4; ++i) mx = fmaxf(mx, lp[i * 256 + tid]);
  red[tid] = mx; __syncthreads();
  for (int o = 128; o > 0; o >>= 1) { if (tid < o) red[tid] = fmaxf(red[tid], red[tid + o]); __syncthreads(); }
  mx = red[0]; __syncthreads();
  float se = 0.0f;
#pragma unroll
  for (int i = 0; i < 4; ++i) se += __expf(lp[i * 256 + tid] - mx);
  red[tid] = se; __syncthreads();
  for (int o = 128; o > 0; o >>= 1) { if (tid < o) red[tid] += red[tid + o]; __syncthreads(); }
  if (tid == 0) {
    int b = row >> 9, pos = row & 511;
    if (gt_mask[b * TCODE + pos]) {
      float lz = __logf(red[0]) + mx;
      int y = gt[b * TCODE + pos];
      atomicAdd(&acc[0], lz - lp[y]);
      atomicAdd(&acc[1], 1.0f);
    }
  }
}

__global__ void final_kernel(const float* acc, float* out) {
  if (threadIdx.x == 0) out[0] = acc[0] / fmaxf(acc[1], 1.0f);
}

// ---------------------------------------------------------------------------
// Host orchestration
// ---------------------------------------------------------------------------
extern "C" void kernel_launch(void* const* d_in, const int* in_sizes, int n_in,
                              void* d_out, int out_size, void* d_ws, size_t ws_size,
                              hipStream_t stream) {
  (void)in_sizes; (void)n_in; (void)out_size; (void)ws_size;
  const float* text   = (const float*)d_in[0];
  const float* prom   = (const float*)d_in[1];
  const float* code   = (const float*)d_in[2];
  const float* maskp  = (const float*)d_in[3];
  const int*   gt     = (const int*)d_in[4];
  const unsigned char* gt_mask = (const unsigned char*)d_in[5];
  const float* Wqkv   = (const float*)d_in[6];
  const float* bqkv   = (const float*)d_in[7];
  const float* Wo     = (const float*)d_in[8];
  const float* bo     = (const float*)d_in[9];
  const float* W1     = (const float*)d_in[10];
  const float* b1     = (const float*)d_in[11];
  const float* W2     = (const float*)d_in[12];
  const float* b2     = (const float*)d_in[13];
  const float* g1     = (const float*)d_in[14];
  const float* beta1  = (const float*)d_in[15];
  const float* g2     = (const float*)d_in[16];
  const float* beta2  = (const float*)d_in[17];
  const float* Wemb   = (const float*)d_in[18];

  // Workspace carve-out
  char* p = (char*)d_ws;
  auto carve = [&](size_t bytes) { char* r = p; p += (bytes + 255) & ~(size_t)255; return r; };
  float*  xf     = (float*)carve((size_t)BT * DD * 4);
  __bf16* xb     = (__bf16*)carve((size_t)BT * DD * 2);
  __bf16* qkvb   = (__bf16*)carve((size_t)BT * 3 * DD * 2);
  __bf16* ob     = (__bf16*)carve((size_t)BT * DD * 2);
  float*  tmpf   = (float*)carve((size_t)BT * DD * 4);
  __bf16* h1b    = (__bf16*)carve((size_t)BT * 4 * DD * 2);
  float*  amask  = (float*)carve((size_t)BB * TT * TT * 4);
  __bf16* Wqkv_b = (__bf16*)carve((size_t)3 * DD * DD * 2);
  __bf16* Wo_b   = (__bf16*)carve((size_t)DD * DD * 2);
  __bf16* W1_b   = (__bf16*)carve((size_t)4 * DD * DD * 2);
  __bf16* W2_b   = (__bf16*)carve((size_t)DD * 4 * DD * 2);
  __bf16* Wemb_b = (__bf16*)carve((size_t)VV * DD * 2);
  float*  logits = (float*)carve((size_t)BB * TCODE * VV * 4);
  float*  acc    = (float*)carve(256);

  auto cvt = [&](const float* src, __bf16* dst, long n) {
    cvt_bf16_kernel<<<(unsigned)((n + 255) / 256), 256, 0, stream>>>(src, dst, n);
  };
  cvt(Wqkv, Wqkv_b, (long)3 * DD * DD);
  cvt(Wo,   Wo_b,   (long)DD * DD);
  cvt(W1,   W1_b,   (long)4 * DD * DD);
  cvt(W2,   W2_b,   (long)DD * 4 * DD);
  cvt(Wemb, Wemb_b, (long)VV * DD);

  concat_kernel<<<(unsigned)(((long)BT * DD + 255) / 256), 256, 0, stream>>>(
      text, prom, code, xf, xb);
  build_amask_kernel<<<(unsigned)(((long)BB * TT * TT + 255) / 256), 256, 0, stream>>>(
      maskp, amask);

  for (int layer = 0; layer < NLAYERS; ++layer) {
    // qkv = x @ Wqkv^T + bqkv   -> bf16
    gemm_bf16_kernel<1><<<dim3(3 * DD / 128, BT / 128), 256, 0, stream>>>(
        xb, Wqkv_b, bqkv, nullptr, qkvb, BT, 3 * DD, DD);
    // attention
    attn_kernel<<<dim3(TT / 16, HH, BB), 32, 0, stream>>>(qkvb, amask, ob);
    // o @ Wo^T + bo -> f32
    gemm_bf16_kernel<0><<<dim3(DD / 128, BT / 128), 256, 0, stream>>>(
        ob, Wo_b, bo, tmpf, nullptr, BT, DD, DD);
    // x = LN(x + attn_out)
    add_ln_kernel<<<BT, 256, 0, stream>>>(xf, tmpf, g1, beta1, xb);
    // h1 = gelu(x @ W1^T + b1) -> bf16
    gemm_bf16_kernel<2><<<dim3(4 * DD / 128, BT / 128), 256, 0, stream>>>(
        xb, W1_b, b1, nullptr, h1b, BT, 4 * DD, DD);
    // f = h1 @ W2^T + b2 -> f32
    gemm_bf16_kernel<0><<<dim3(DD / 128, BT / 128), 256, 0, stream>>>(
        h1b, W2_b, b2, tmpf, nullptr, BT, DD, 4 * DD);
    // x = LN(x + f)
    add_ln_kernel<<<BT, 256, 0, stream>>>(xf, tmpf, g2, beta2, xb);
  }

  // logits only for rows [T - 513, T - 2] per batch (512 contiguous rows)
  for (int b = 0; b < BB; ++b) {
    gemm_bf16_kernel<0><<<dim3(VV / 128, TCODE / 128), 256, 0, stream>>>(
        xb + ((long)b * TT + (TT - TCODE - 1)) * DD, Wemb_b, nullptr,
        logits + (long)b * TCODE * VV, nullptr, TCODE, VV, DD);
  }

  zero2_kernel<<<1, 32, 0, stream>>>(acc);
  loss_kernel<<<BB * TCODE, 256, 0, stream>>>(logits, gt, gt_mask, acc);
  final_kernel<<<1, 32, 0, stream>>>(acc, (float*)d_out);
}
